// Model_25881472926517
// MI455X (gfx1250) — matrix operand order repaired and checked by reading, not compiled, not run on hardware
//
#include <hip/hip_runtime.h>
#include <hip/hip_bf16.h>

// ---------------- model constants ----------------
#define NN      2048
#define BB      16
#define TIN     24
#define HOR     24
#define RNNH    16
#define MEMN    20
#define MEMD    16
#define DECH    32
#define C_ENC   17            // IN_DIM + RNN
#define C_DEC   34            // DEC_IN + DEC_HID
#define F_ENC   (BB*C_ENC)    // 272
#define FP_ENC  288           // padded to 32-multiple
#define F_DEC   (BB*C_DEC)    // 544 (already 17*32)
#define FP_DEC  544

typedef __attribute__((ext_vector_type(16))) __bf16 bf16x16;
typedef __attribute__((ext_vector_type(8)))  float  f32x8;

union FragU { struct { uint4 lo, hi; } u; bf16x16 v; };

__device__ __forceinline__ float sigf(float v) { return 1.0f / (1.0f + expf(-v)); }

// ---------------- e1/e2 = We @ Memory ----------------
__global__ void k_embed(const float* __restrict__ We1, const float* __restrict__ We2,
                        const float* __restrict__ Mem, float* __restrict__ e1,
                        float* __restrict__ e2) {
  int n = blockIdx.x;
  int t = threadIdx.x;           // 0..31
  const float* We = (t < 16) ? We1 : We2;
  float* eo = (t < 16) ? e1 : e2;
  int d = t & 15;
  float acc = 0.f;
  #pragma unroll
  for (int j = 0; j < MEMN; j++) acc += We[n * MEMN + j] * Mem[j * MEMD + d];
  eo[n * MEMD + d] = acc;
}

// ---------------- g = softmax(relu(ea @ eb^T)) row-wise; writes g (bf16) and g^T ----------------
__global__ __launch_bounds__(256) void k_graph(const float* __restrict__ e1,
                                               const float* __restrict__ e2,
                                               __hip_bfloat16* __restrict__ bases,
                                               __hip_bfloat16* __restrict__ g1T,
                                               __hip_bfloat16* __restrict__ g2T) {
  int row = blockIdx.x;
  int which = blockIdx.y;        // 0 -> g1, 1 -> g2
  const float* ea = which ? e2 : e1;
  const float* eb = which ? e1 : e2;
  __hip_bfloat16* g  = bases + (size_t)(which * 2) * NN * NN;
  __hip_bfloat16* gT = which ? g2T : g1T;
  int tid = threadIdx.x;

  float a[MEMD];
  #pragma unroll
  for (int d = 0; d < MEMD; d++) a[d] = ea[row * MEMD + d];

  float s[8];
  float mx = -1e30f;
  #pragma unroll
  for (int i = 0; i < 8; i++) {
    int j = tid + i * 256;
    const float* b = eb + (size_t)j * MEMD;
    float acc = 0.f;
    #pragma unroll
    for (int d = 0; d < MEMD; d++) acc += a[d] * b[d];
    acc = fmaxf(acc, 0.f);
    s[i] = acc;
    mx = fmaxf(mx, acc);
  }
  __shared__ float red[256];
  red[tid] = mx; __syncthreads();
  for (int off = 128; off > 0; off >>= 1) {
    if (tid < off) red[tid] = fmaxf(red[tid], red[tid + off]);
    __syncthreads();
  }
  mx = red[0]; __syncthreads();
  float sum = 0.f;
  #pragma unroll
  for (int i = 0; i < 8; i++) { s[i] = expf(s[i] - mx); sum += s[i]; }
  red[tid] = sum; __syncthreads();
  for (int off = 128; off > 0; off >>= 1) {
    if (tid < off) red[tid] += red[tid + off];
    __syncthreads();
  }
  float inv = 1.0f / red[0];
  #pragma unroll
  for (int i = 0; i < 8; i++) {
    int j = tid + i * 256;
    float v = s[i] * inv;
    g[(size_t)row * NN + j]  = __float2bfloat16(v);
    gT[(size_t)j * NN + row] = __float2bfloat16(v);
  }
}

// ---------------- WMMA GEMM: out = A(2048xK,bf16,row-major) @ X, X given transposed (FxK) ----
// mode 0: Y(f32, ldy) = acc, per blockIdx.z base
// mode 1: Yb(bf16, ld=NN) = 2*acc - I   (Chebyshev second basis)
__global__ __launch_bounds__(256) void k_gemm(const __hip_bfloat16* __restrict__ A_,
                                              long long strideAz,
                                              const __hip_bfloat16* __restrict__ XT_,
                                              float* __restrict__ Y, long long strideYz,
                                              int ldy, __hip_bfloat16* __restrict__ Yb,
                                              int K, int mode) {
  const __bf16* A  = reinterpret_cast<const __bf16*>(A_) + (long long)blockIdx.z * strideAz;
  const __bf16* XT = reinterpret_cast<const __bf16*>(XT_);
  int lane = threadIdx.x & 31;
  int wave = threadIdx.x >> 5;
  int half = lane >> 4;
  int lr   = lane & 15;
  int fc = blockIdx.x * 32;                    // 2 column tiles
  int mr = blockIdx.y * 512 + wave * 64;       // 4 row tiles per wave

  f32x8 zero8 = {0.f, 0.f, 0.f, 0.f, 0.f, 0.f, 0.f, 0.f};
  f32x8 acc[4][2];
  #pragma unroll
  for (int i = 0; i < 4; i++)
    #pragma unroll
    for (int j = 0; j < 2; j++) acc[i][j] = zero8;

  for (int kc = 0; kc < K; kc += 32) {
    FragU bfr[2];
    #pragma unroll
    for (int j = 0; j < 2; j++) {
      // B fragment, this code's k order: lanes 0-15 hold K=kc..kc+15, lanes 16-31 hold kc+16..kc+31
      const __bf16* p = XT + (size_t)(fc + j * 16 + lr) * K + (kc + half * 16);
      bfr[j].u.lo = *reinterpret_cast<const uint4*>(p);
      bfr[j].u.hi = *reinterpret_cast<const uint4*>(p + 8);
    }
    FragU afr[4];
    #pragma unroll
    for (int i = 0; i < 4; i++) {
      // A fragment, this code's k order (same as B): lanes 0-15 hold K 0..15, lanes 16-31 hold K 16..31
      const __bf16* p = A + (size_t)(mr + i * 16 + lr) * K + (kc + half * 16);
      afr[i].u.lo = *reinterpret_cast<const uint4*>(p);
      afr[i].u.hi = *reinterpret_cast<const uint4*>(p + 8);
      __builtin_prefetch(p + 128, 0, 1);       // global_prefetch_b8, 4 K-chunks ahead
    }
    #pragma unroll
    for (int i = 0; i < 4; i++)
      #pragma unroll
      for (int j = 0; j < 2; j++)
        acc[i][j] = __builtin_amdgcn_wmma_f32_16x16x32_bf16(
            false, afr[i].v, false, bfr[j].v, (short)0, acc[i][j], false, false);
  }

  if (mode == 0) {
    float* Yz = Y + (long long)blockIdx.z * strideYz;
    #pragma unroll
    for (int i = 0; i < 4; i++)
      for (int j = 0; j < 2; j++)
        #pragma unroll
        for (int v = 0; v < 8; v++) {
          int row = mr + i * 16 + half * 8 + v;
          int col = fc + j * 16 + lr;
          Yz[(size_t)row * ldy + col] = acc[i][j][v];
        }
  } else {
    #pragma unroll
    for (int i = 0; i < 4; i++)
      for (int j = 0; j < 2; j++)
        #pragma unroll
        for (int v = 0; v < 8; v++) {
          int row = mr + i * 16 + half * 8 + v;
          int col = fc + j * 16 + lr;
          float val = 2.0f * acc[i][j][v] - (row == col ? 1.0f : 0.0f);
          Yb[(size_t)row * NN + col] = __float2bfloat16(val);
        }
  }
}

// ---------------- utility ----------------
__global__ void k_zero(float* p, int n) {
  int i = blockIdx.x * 256 + threadIdx.x;
  if (i < n) p[i] = 0.f;
}

// ---------------- encoder: build transposed input (FP_ENC x NN, bf16); zero pad rows of both XT bufs
__global__ void k_build_enc_x1(const float* __restrict__ x, const float* __restrict__ h,
                               __hip_bfloat16* __restrict__ XT1,
                               __hip_bfloat16* __restrict__ XT2, int t) {
  int idx = blockIdx.x * 256 + threadIdx.x;
  if (idx >= FP_ENC * NN) return;
  int f = idx >> 11, m = idx & (NN - 1);
  if (f >= F_ENC) {
    XT1[idx] = __float2bfloat16(0.f);
    XT2[idx] = __float2bfloat16(0.f);
    return;
  }
  int b = f / C_ENC, c = f % C_ENC;
  float v = (c == 0) ? x[((size_t)(b * TIN + t)) * NN + m]
                     : h[((size_t)b * NN + m) * RNNH + (c - 1)];
  XT1[idx] = __float2bfloat16(v);
}

// ---------------- encoder gate: z_r = sigmoid(x_g @ W + b); emit r, z*h, cand XT2 ----------------
__global__ __launch_bounds__(256) void k_enc_gate(const float* __restrict__ x,
                                                  const float* __restrict__ h,
                                                  const float* __restrict__ Y,
                                                  const float* __restrict__ W,
                                                  const float* __restrict__ bias,
                                                  float* __restrict__ r_buf,
                                                  float* __restrict__ zh_buf,
                                                  __hip_bfloat16* __restrict__ XT2, int t) {
  __shared__ float Ws[6 * C_ENC * 2 * RNNH];
  for (int i = threadIdx.x; i < 6 * C_ENC * 2 * RNNH; i += 256) Ws[i] = W[i];
  __syncthreads();
  int idx = blockIdx.x * 256 + threadIdx.x;
  int b = idx >> 11, n = idx & (NN - 1);
  float hv[RNNH];
  #pragma unroll
  for (int j = 0; j < RNNH; j++) hv[j] = h[((size_t)b * NN + n) * RNNH + j];
  float x0 = x[((size_t)(b * TIN + t)) * NN + n];

  float zr[2 * RNNH];
  #pragma unroll
  for (int o = 0; o < 2 * RNNH; o++) zr[o] = bias[o];
  // identity bases k=0 and k=3
  for (int c = 0; c < C_ENC; c++) {
    float xv = (c == 0) ? x0 : hv[c - 1];
    const float* w0 = &Ws[c * 2 * RNNH];
    const float* w3 = &Ws[(3 * C_ENC + c) * 2 * RNNH];
    for (int o = 0; o < 2 * RNNH; o++) zr[o] += xv * (w0[o] + w3[o]);
  }
  const int kkmap[4] = {1, 2, 4, 5};
  for (int z = 0; z < 4; z++) {
    const float* yrow = Y + (size_t)z * NN * FP_ENC + (size_t)n * FP_ENC + b * C_ENC;
    int base = kkmap[z] * C_ENC;
    for (int c = 0; c < C_ENC; c++) {
      float xv = yrow[c];
      const float* w = &Ws[(base + c) * 2 * RNNH];
      for (int o = 0; o < 2 * RNNH; o++) zr[o] += xv * w[o];
    }
  }
  XT2[(size_t)(b * C_ENC) * NN + n] = __float2bfloat16(x0);
  #pragma unroll
  for (int j = 0; j < RNNH; j++) {
    float zz = sigf(zr[j]);
    float rr = sigf(zr[RNNH + j]);
    float zh = zz * hv[j];
    r_buf[((size_t)b * NN + n) * RNNH + j] = rr;
    zh_buf[((size_t)b * NN + n) * RNNH + j] = zh;
    XT2[(size_t)(b * C_ENC + 1 + j) * NN + n] = __float2bfloat16(zh);
  }
}

// ---------------- encoder update: hc = tanh(...); h = r*h + (1-r)*hc ----------------
__global__ __launch_bounds__(256) void k_enc_up(const float* __restrict__ x,
                                                const float* __restrict__ zh_buf,
                                                const float* __restrict__ r_buf,
                                                const float* __restrict__ Y,
                                                const float* __restrict__ W,
                                                const float* __restrict__ bias,
                                                float* h, int t) {
  __shared__ float Ws[6 * C_ENC * RNNH];
  for (int i = threadIdx.x; i < 6 * C_ENC * RNNH; i += 256) Ws[i] = W[i];
  __syncthreads();
  int idx = blockIdx.x * 256 + threadIdx.x;
  int b = idx >> 11, n = idx & (NN - 1);
  float x0 = x[((size_t)(b * TIN + t)) * NN + n];
  float zh[RNNH];
  #pragma unroll
  for (int j = 0; j < RNNH; j++) zh[j] = zh_buf[((size_t)b * NN + n) * RNNH + j];

  float hc[RNNH];
  #pragma unroll
  for (int o = 0; o < RNNH; o++) hc[o] = bias[o];
  for (int c = 0; c < C_ENC; c++) {
    float xv = (c == 0) ? x0 : zh[c - 1];
    const float* w0 = &Ws[c * RNNH];
    const float* w3 = &Ws[(3 * C_ENC + c) * RNNH];
    for (int o = 0; o < RNNH; o++) hc[o] += xv * (w0[o] + w3[o]);
  }
  const int kkmap[4] = {1, 2, 4, 5};
  for (int z = 0; z < 4; z++) {
    const float* yrow = Y + (size_t)z * NN * FP_ENC + (size_t)n * FP_ENC + b * C_ENC;
    int base = kkmap[z] * C_ENC;
    for (int c = 0; c < C_ENC; c++) {
      float xv = yrow[c];
      const float* w = &Ws[(base + c) * RNNH];
      for (int o = 0; o < RNNH; o++) hc[o] += xv * w[o];
    }
  }
  #pragma unroll
  for (int j = 0; j < RNNH; j++) {
    size_t hi = ((size_t)b * NN + n) * RNNH + j;
    float rr = r_buf[hi];
    float hcv = tanhf(hc[j]);
    h[hi] = rr * h[hi] + (1.f - rr) * hcv;
  }
}

// ---------------- attention: h_d = [h, softmax(hWq M^T) M]; go = 0 ----------------
__global__ __launch_bounds__(256) void k_attention(const float* __restrict__ h,
                                                   const float* __restrict__ Wq,
                                                   const float* __restrict__ Mem,
                                                   float* __restrict__ hdec,
                                                   float* __restrict__ go) {
  __shared__ float Wqs[RNNH * MEMD];
  __shared__ float Ms[MEMN * MEMD];
  for (int i = threadIdx.x; i < RNNH * MEMD; i += 256) Wqs[i] = Wq[i];
  for (int i = threadIdx.x; i < MEMN * MEMD; i += 256) Ms[i] = Mem[i];
  __syncthreads();
  int idx = blockIdx.x * 256 + threadIdx.x;
  int b = idx >> 11, n = idx & (NN - 1);
  float hv[RNNH];
  #pragma unroll
  for (int j = 0; j < RNNH; j++) hv[j] = h[((size_t)b * NN + n) * RNNH + j];
  float q[MEMD];
  #pragma unroll
  for (int d = 0; d < MEMD; d++) {
    float a = 0.f;
    #pragma unroll
    for (int j = 0; j < RNNH; j++) a += hv[j] * Wqs[j * MEMD + d];
    q[d] = a;
  }
  float sc[MEMN];
  float mx = -1e30f;
  for (int m = 0; m < MEMN; m++) {
    float a = 0.f;
    #pragma unroll
    for (int d = 0; d < MEMD; d++) a += q[d] * Ms[m * MEMD + d];
    sc[m] = a;
    mx = fmaxf(mx, a);
  }
  float sum = 0.f;
  for (int m = 0; m < MEMN; m++) { sc[m] = expf(sc[m] - mx); sum += sc[m]; }
  float inv = 1.f / sum;
  #pragma unroll
  for (int d = 0; d < MEMD; d++) {
    float v = 0.f;
    for (int m = 0; m < MEMN; m++) v += sc[m] * inv * Ms[m * MEMD + d];
    hdec[((size_t)b * NN + n) * DECH + RNNH + d] = v;
  }
  #pragma unroll
  for (int j = 0; j < RNNH; j++) hdec[((size_t)b * NN + n) * DECH + j] = hv[j];
  go[(size_t)b * NN + n] = 0.f;
}

// ---------------- decoder build XT1 ----------------
__global__ void k_build_dec_x1(const float* __restrict__ go, const float* __restrict__ h,
                               __hip_bfloat16* __restrict__ XT1) {
  int idx = blockIdx.x * 256 + threadIdx.x;
  if (idx >= F_DEC * NN) return;
  int f = idx >> 11, m = idx & (NN - 1);
  int b = f / C_DEC, c = f % C_DEC;
  float v = (c == 0) ? go[(size_t)b * NN + m]
          : (c == 1) ? 0.f
                     : h[((size_t)b * NN + m) * DECH + (c - 2)];
  XT1[idx] = __float2bfloat16(v);
}

// ---------------- decoder gate ----------------
__global__ __launch_bounds__(256) void k_dec_gate(const float* __restrict__ go,
                                                  const float* __restrict__ h,
                                                  const float* __restrict__ Y,
                                                  const float* __restrict__ W,
                                                  const float* __restrict__ bias,
                                                  float* __restrict__ r_buf,
                                                  float* __restrict__ zh_buf,
                                                  __hip_bfloat16* __restrict__ XT2) {
  __shared__ float Ws[6 * C_DEC * 2 * DECH];   // 204*64 floats = 52KB
  for (int i = threadIdx.x; i < 6 * C_DEC * 2 * DECH; i += 256) Ws[i] = W[i];
  __syncthreads();
  int idx = blockIdx.x * 256 + threadIdx.x;
  int b = idx >> 11, n = idx & (NN - 1);
  float hv[DECH];
  #pragma unroll
  for (int j = 0; j < DECH; j++) hv[j] = h[((size_t)b * NN + n) * DECH + j];
  float g0 = go[(size_t)b * NN + n];

  float zr[2 * DECH];
  #pragma unroll
  for (int o = 0; o < 2 * DECH; o++) zr[o] = bias[o];
  for (int c = 0; c < C_DEC; c++) {
    float xv = (c == 0) ? g0 : (c == 1) ? 0.f : hv[c - 2];
    const float* w0 = &Ws[c * 2 * DECH];
    const float* w3 = &Ws[(3 * C_DEC + c) * 2 * DECH];
    for (int o = 0; o < 2 * DECH; o++) zr[o] += xv * (w0[o] + w3[o]);
  }
  const int kkmap[4] = {1, 2, 4, 5};
  for (int z = 0; z < 4; z++) {
    const float* yrow = Y + (size_t)z * NN * FP_DEC + (size_t)n * FP_DEC + b * C_DEC;
    int base = kkmap[z] * C_DEC;
    for (int c = 0; c < C_DEC; c++) {
      float xv = yrow[c];
      const float* w = &Ws[(base + c) * 2 * DECH];
      for (int o = 0; o < 2 * DECH; o++) zr[o] += xv * w[o];
    }
  }
  XT2[(size_t)(b * C_DEC + 0) * NN + n] = __float2bfloat16(g0);
  XT2[(size_t)(b * C_DEC + 1) * NN + n] = __float2bfloat16(0.f);
  #pragma unroll
  for (int j = 0; j < DECH; j++) {
    float zz = sigf(zr[j]);
    float rr = sigf(zr[DECH + j]);
    float zh = zz * hv[j];
    r_buf[((size_t)b * NN + n) * DECH + j] = rr;
    zh_buf[((size_t)b * NN + n) * DECH + j] = zh;
    XT2[(size_t)(b * C_DEC + 2 + j) * NN + n] = __float2bfloat16(zh);
  }
}

// ---------------- decoder update + projection + output ----------------
__global__ __launch_bounds__(256) void k_dec_up(const float* __restrict__ zh_buf,
                                                const float* __restrict__ r_buf,
                                                const float* __restrict__ Y,
                                                const float* __restrict__ W,
                                                const float* __restrict__ bias,
                                                const float* __restrict__ pW,
                                                const float* __restrict__ pB,
                                                float* h, float* go_buf,
                                                float* __restrict__ out, int t) {
  __shared__ float Ws[6 * C_DEC * DECH];       // 204*32 floats
  for (int i = threadIdx.x; i < 6 * C_DEC * DECH; i += 256) Ws[i] = W[i];
  __syncthreads();
  int idx = blockIdx.x * 256 + threadIdx.x;
  int b = idx >> 11, n = idx & (NN - 1);
  float g0 = go_buf[(size_t)b * NN + n];
  float zh[DECH];
  #pragma unroll
  for (int j = 0; j < DECH; j++) zh[j] = zh_buf[((size_t)b * NN + n) * DECH + j];

  float hc[DECH];
  #pragma unroll
  for (int o = 0; o < DECH; o++) hc[o] = bias[o];
  for (int c = 0; c < C_DEC; c++) {
    float xv = (c == 0) ? g0 : (c == 1) ? 0.f : zh[c - 2];
    const float* w0 = &Ws[c * DECH];
    const float* w3 = &Ws[(3 * C_DEC + c) * DECH];
    for (int o = 0; o < DECH; o++) hc[o] += xv * (w0[o] + w3[o]);
  }
  const int kkmap[4] = {1, 2, 4, 5};
  for (int z = 0; z < 4; z++) {
    const float* yrow = Y + (size_t)z * NN * FP_DEC + (size_t)n * FP_DEC + b * C_DEC;
    int base = kkmap[z] * C_DEC;
    for (int c = 0; c < C_DEC; c++) {
      float xv = yrow[c];
      const float* w = &Ws[(base + c) * DECH];
      for (int o = 0; o < DECH; o++) hc[o] += xv * w[o];
    }
  }
  float gacc = pB[0];
  #pragma unroll
  for (int j = 0; j < DECH; j++) {
    size_t hi = ((size_t)b * NN + n) * DECH + j;
    float rr = r_buf[hi];
    float hn = rr * h[hi] + (1.f - rr) * tanhf(hc[j]);
    h[hi] = hn;
    gacc += hn * pW[j];
  }
  go_buf[(size_t)b * NN + n] = gacc;
  out[((size_t)b * HOR + t) * NN + n] = gacc;
}

// ---------------- host ----------------
extern "C" void kernel_launch(void* const* d_in, const int* in_sizes, int n_in,
                              void* d_out, int out_size, void* d_ws, size_t ws_size,
                              hipStream_t stream) {
  const float* x   = (const float*)d_in[0];
  const float* Mem = (const float*)d_in[1];
  const float* Wq  = (const float*)d_in[2];
  const float* We1 = (const float*)d_in[3];
  const float* We2 = (const float*)d_in[4];
  const float* egW = (const float*)d_in[5];
  const float* egB = (const float*)d_in[6];
  const float* euW = (const float*)d_in[7];
  const float* euB = (const float*)d_in[8];
  const float* dgW = (const float*)d_in[9];
  const float* dgB = (const float*)d_in[10];
  const float* duW = (const float*)d_in[11];
  const float* duB = (const float*)d_in[12];
  const float* pW  = (const float*)d_in[13];
  const float* pB  = (const float*)d_in[14];
  float* out = (float*)d_out;

  char* ws = (char*)d_ws;
  size_t off = 0;
  auto take = [&](size_t bytes) -> char* {
    char* p = ws + off;
    off += (bytes + 255) & ~(size_t)255;
    return p;
  };
  const size_t SZ = (size_t)NN * NN;
  float* e1 = (float*)take((size_t)NN * MEMD * 4);
  float* e2 = (float*)take((size_t)NN * MEMD * 4);
  __hip_bfloat16* bases = (__hip_bfloat16*)take(4 * SZ * 2);      // g1, cheb(g1), g2, cheb(g2)
  __hip_bfloat16* XT1   = (__hip_bfloat16*)take((size_t)FP_DEC * NN * 2);
  __hip_bfloat16* XT2   = (__hip_bfloat16*)take((size_t)FP_DEC * NN * 2);
  float* h_enc  = (float*)take((size_t)BB * NN * RNNH * 4);
  float* h_dec  = (float*)take((size_t)BB * NN * DECH * 4);
  float* r_buf  = (float*)take((size_t)BB * NN * DECH * 4);
  float* zh_buf = (float*)take((size_t)BB * NN * DECH * 4);
  float* go_buf = (float*)take((size_t)BB * NN * 4);
  // gT buffers (used only before the recurrent steps) overlaid with conv-output Y
  size_t gt_bytes = 2 * SZ * 2;
  size_t y_bytes  = (size_t)4 * NN * FP_DEC * 4;
  char* shared_region = take(gt_bytes > y_bytes ? gt_bytes : y_bytes);
  __hip_bfloat16* g1T = (__hip_bfloat16*)shared_region;
  __hip_bfloat16* g2T = (__hip_bfloat16*)(shared_region + SZ * 2);
  float* Yb = (float*)shared_region;

  // ---- graph construction ----
  k_embed<<<NN, 32, 0, stream>>>(We1, We2, Mem, e1, e2);
  k_graph<<<dim3(NN, 2), 256, 0, stream>>>(e1, e2, bases, g1T, g2T);
  // cheb: bases[1] = 2*g1@g1 - I ; bases[3] = 2*g2@g2 - I
  k_gemm<<<dim3(NN / 32, NN / 512, 1), 256, 0, stream>>>(
      bases + 0 * SZ, 0, g1T, nullptr, 0, 0, bases + 1 * SZ, NN, 1);
  k_gemm<<<dim3(NN / 32, NN / 512, 1), 256, 0, stream>>>(
      bases + 2 * SZ, 0, g2T, nullptr, 0, 0, bases + 3 * SZ, NN, 1);

  // ---- encoder ----
  k_zero<<<(BB * NN * RNNH + 255) / 256, 256, 0, stream>>>(h_enc, BB * NN * RNNH);
  for (int t = 0; t < TIN; t++) {
    k_build_enc_x1<<<(FP_ENC * NN) / 256, 256, 0, stream>>>(x, h_enc, XT1, XT2, t);
    k_gemm<<<dim3(FP_ENC / 32, NN / 512, 4), 256, 0, stream>>>(
        bases, (long long)SZ, XT1, Yb, (long long)NN * FP_ENC, FP_ENC, nullptr, NN, 0);
    k_enc_gate<<<(BB * NN) / 256, 256, 0, stream>>>(x, h_enc, Yb, egW, egB, r_buf, zh_buf, XT2, t);
    k_gemm<<<dim3(FP_ENC / 32, NN / 512, 4), 256, 0, stream>>>(
        bases, (long long)SZ, XT2, Yb, (long long)NN * FP_ENC, FP_ENC, nullptr, NN, 0);
    k_enc_up<<<(BB * NN) / 256, 256, 0, stream>>>(x, zh_buf, r_buf, Yb, euW, euB, h_enc, t);
  }

  // ---- attention / decoder init ----
  k_attention<<<(BB * NN) / 256, 256, 0, stream>>>(h_enc, Wq, Mem, h_dec, go_buf);

  // ---- decoder ----
  for (int t = 0; t < HOR; t++) {
    k_build_dec_x1<<<(F_DEC * NN) / 256, 256, 0, stream>>>(go_buf, h_dec, XT1);
    k_gemm<<<dim3(FP_DEC / 32, NN / 512, 4), 256, 0, stream>>>(
        bases, (long long)SZ, XT1, Yb, (long long)NN * FP_DEC, FP_DEC, nullptr, NN, 0);
    k_dec_gate<<<(BB * NN) / 256, 256, 0, stream>>>(go_buf, h_dec, Yb, dgW, dgB, r_buf, zh_buf, XT2);
    k_gemm<<<dim3(FP_DEC / 32, NN / 512, 4), 256, 0, stream>>>(
        bases, (long long)SZ, XT2, Yb, (long long)NN * FP_DEC, FP_DEC, nullptr, NN, 0);
    k_dec_up<<<(BB * NN) / 256, 256, 0, stream>>>(zh_buf, r_buf, Yb, duW, duB, pW, pB,
                                                  h_dec, go_buf, out, t);
  }
}
